// EncoderRNN_20375324852351
// MI455X (gfx1250) — matrix operand order repaired and checked by reading, not compiled, not run on hardware
//
#include <hip/hip_runtime.h>
#include <hip/hip_bf16.h>

typedef __attribute__((ext_vector_type(16))) _Float16 v16h;
typedef __attribute__((ext_vector_type(8)))  _Float16 v8h;
typedef __attribute__((ext_vector_type(8)))  float    v8f;

#define NROWS 2048      // T*B
#define DIM   1000      // D == H
#define KP    1024      // padded K
#define GATES 4000      // 4*H
#define NLAYER 4

// -------- f32 -> f16 convert with zero-pad of K 1000 -> 1024 --------
__global__ __launch_bounds__(256) void cvt_pad_f16(const float* __restrict__ src,
                                                   _Float16* __restrict__ dst,
                                                   int rows) {
    int idx = blockIdx.x * blockDim.x + threadIdx.x;
    if (idx >= rows * KP) return;
    int r = idx >> 10;          // / KP
    int k = idx & (KP - 1);
    dst[idx] = (k < DIM) ? (_Float16)src[r * DIM + k] : (_Float16)0.0f;
}

__global__ __launch_bounds__(256) void zero_u32(unsigned int* __restrict__ p, int n) {
    int idx = blockIdx.x * blockDim.x + threadIdx.x;
    if (idx < n) p[idx] = 0u;
}

__device__ __forceinline__ float sigmoidf_(float x) {
    return 1.0f / (1.0f + __expf(-x));
}

// -------- one LSTM layer: gates GEMM (WMMA) + fused cell epilogue --------
// grid = (NROWS/128, 32), block = 256 (8 waves: 4 M-rows x 2 J-cols)
// Wave tile: 32 rows x 16 j, four gate subtiles -> i/f/g/o in-lane.
__global__ __launch_bounds__(256)
void lstm_layer_wmma(const _Float16* __restrict__ x16,    // [NROWS,KP] always layer input
                     const _Float16* __restrict__ hin16,  // [NROWS,KP] or unused (first)
                     const _Float16* __restrict__ wih,    // [GATES,KP]
                     const _Float16* __restrict__ whh,    // [GATES,KP]
                     const float* __restrict__ bih,       // [GATES]
                     const float* __restrict__ bhh,       // [GATES]
                     float* __restrict__ cbuf,            // [NROWS,DIM] f32 state
                     _Float16* __restrict__ hout16,       // [NROWS,KP] (if !last)
                     float* __restrict__ out,             // [NROWS,DIM] (if last)
                     int first, int last) {
    const int lane = threadIdx.x & 31;
    const int wave = threadIdx.x >> 5;
    const int wr   = wave & 3;          // wave row 0..3
    const int wc   = wave >> 2;         // wave col 0..1
    const int lh   = lane >> 4;         // lane half
    const int l15  = lane & 15;

    const int m0  = blockIdx.x * 128 + wr * 32;           // wave's first row
    const int j   = blockIdx.y * 32 + wc * 16 + l15;      // this lane's hidden unit
    const int jc  = (j < DIM) ? j : (DIM - 1);            // clamp for loads
    const bool jok = (j < DIM);

    // A fragment element offsets (ld = KP): two 16-row subtiles
    const int aoff0 = (m0 + l15) * KP + lh * 16;
    const int aoff1 = aoff0 + 16 * KP;
    // B fragment offsets: one per gate, contiguous 16 halves along K
    const int kb = lh * 16;
    int boff[4];
#pragma unroll
    for (int g = 0; g < 4; ++g) boff[g] = (g * DIM + jc) * KP + kb;

    v8f acc[2][4];
#pragma unroll
    for (int mi = 0; mi < 2; ++mi)
#pragma unroll
        for (int g = 0; g < 4; ++g)
#pragma unroll
            for (int r = 0; r < 8; ++r) acc[mi][g][r] = 0.0f;

    const int npass = first ? 1 : 2;    // layer 0: h == 0, skip Whh pass
    for (int p = 0; p < npass; ++p) {
        const _Float16* __restrict__ A = (p == 0) ? x16 : hin16;
        const _Float16* __restrict__ W = (p == 0) ? wih : whh;
#pragma unroll 4
        for (int k = 0; k < KP; k += 32) {
            v16h a[2];
            {
                union { v16h v; v8h h[2]; } u;
                u.h[0] = *(const v8h*)(A + aoff0 + k);
                u.h[1] = *(const v8h*)(A + aoff0 + k + 8);
                a[0] = u.v;
                u.h[0] = *(const v8h*)(A + aoff1 + k);
                u.h[1] = *(const v8h*)(A + aoff1 + k + 8);
                a[1] = u.v;
            }
            v16h b[4];
#pragma unroll
            for (int g = 0; g < 4; ++g)
                b[g] = *(const v16h*)(W + boff[g] + k);
#pragma unroll
            for (int mi = 0; mi < 2; ++mi)
#pragma unroll
                for (int g = 0; g < 4; ++g)
                    acc[mi][g] = __builtin_amdgcn_wmma_f32_16x16x32_f16(
                        false, a[mi], false, b[g], (short)0, acc[mi][g],
                        false, false);
        }
    }

    // ---- fused LSTM cell epilogue: i/f/g/o are in-lane ----
    float bsum[4];
#pragma unroll
    for (int g = 0; g < 4; ++g)
        bsum[g] = jok ? (bih[g * DIM + j] + bhh[g * DIM + j]) : 0.0f;

#pragma unroll
    for (int mi = 0; mi < 2; ++mi) {
#pragma unroll
        for (int r = 0; r < 8; ++r) {
            const int row = m0 + mi * 16 + lh * 8 + r;   // C layout: M = r + 8*laneHalf
            float iv = acc[mi][0][r] + bsum[0];
            float fv = acc[mi][1][r] + bsum[1];
            float gv = acc[mi][2][r] + bsum[2];
            float ov = acc[mi][3][r] + bsum[3];
            if (jok) {
                float si = sigmoidf_(iv);
                float sf = sigmoidf_(fv);
                float so = sigmoidf_(ov);
                float tg = tanhf(gv);
                float cold = first ? 0.0f : cbuf[row * DIM + j];
                float cn = sf * cold + si * tg;
                float hn = so * tanhf(cn);
                cbuf[row * DIM + j] = cn;
                if (last) out[row * DIM + j] = fmaxf(hn, 0.0f);
                else      hout16[row * KP + j] = (_Float16)hn;
            }
        }
    }
}

extern "C" void kernel_launch(void* const* d_in, const int* in_sizes, int n_in,
                              void* d_out, int out_size, void* d_ws, size_t ws_size,
                              hipStream_t stream) {
    (void)in_sizes; (void)n_in; (void)out_size; (void)ws_size;
    const float* x   = (const float*)d_in[0];   // [2048,1,1000]
    const float* wih = (const float*)d_in[1];   // [4,4000,1000]
    const float* whh = (const float*)d_in[2];   // [4,4000,1000]
    const float* bih = (const float*)d_in[3];   // [4,4000]
    const float* bhh = (const float*)d_in[4];   // [4,4000]
    float* out = (float*)d_out;                 // [2048,1,1000]

    char* ws = (char*)d_ws;
    size_t off = 0;
    auto alloc = [&](size_t bytes) {
        void* p = ws + off;
        off = (off + bytes + 255) & ~(size_t)255;
        return p;
    };
    _Float16* x16   = (_Float16*)alloc((size_t)NROWS * KP * 2);
    _Float16* h16a  = (_Float16*)alloc((size_t)NROWS * KP * 2);
    _Float16* h16b  = (_Float16*)alloc((size_t)NROWS * KP * 2);
    _Float16* wih16 = (_Float16*)alloc((size_t)NLAYER * GATES * KP * 2);
    _Float16* whh16 = (_Float16*)alloc((size_t)NLAYER * GATES * KP * 2);
    float*    cbuf  = (float*)alloc((size_t)NROWS * DIM * 4);

    // convert + zero-pad inputs/weights to f16 [.,1024]
    {
        int total = NROWS * KP;
        cvt_pad_f16<<<(total + 255) / 256, 256, 0, stream>>>(x, x16, NROWS);
    }
    {
        int rows  = NLAYER * GATES;
        int total = rows * KP;
        cvt_pad_f16<<<(total + 255) / 256, 256, 0, stream>>>(wih, wih16, rows);
        cvt_pad_f16<<<(total + 255) / 256, 256, 0, stream>>>(whh, whh16, rows);
    }
    // zero h ping-pong buffers (pad columns must be 0 for the Whh GEMM)
    {
        int n = (NROWS * KP * 2) / 4;   // u32 count per buffer
        zero_u32<<<(n + 255) / 256, 256, 0, stream>>>((unsigned int*)h16a, n);
        zero_u32<<<(n + 255) / 256, 256, 0, stream>>>((unsigned int*)h16b, n);
    }

    dim3 grid(NROWS / 128, 32);   // (16, 32); j-block 31 partially masked (1000..1023)
    const size_t WL = (size_t)GATES * KP;   // f16 elements per layer weight

    // layer 0: h=0 -> single GEMM pass, c_old = 0
    lstm_layer_wmma<<<grid, 256, 0, stream>>>(
        x16, h16a, wih16 + 0 * WL, whh16 + 0 * WL,
        bih + 0 * GATES, bhh + 0 * GATES, cbuf, h16a, out, 1, 0);
    // layer 1
    lstm_layer_wmma<<<grid, 256, 0, stream>>>(
        x16, h16a, wih16 + 1 * WL, whh16 + 1 * WL,
        bih + 1 * GATES, bhh + 1 * GATES, cbuf, h16b, out, 0, 0);
    // layer 2
    lstm_layer_wmma<<<grid, 256, 0, stream>>>(
        x16, h16b, wih16 + 2 * WL, whh16 + 2 * WL,
        bih + 2 * GATES, bhh + 2 * GATES, cbuf, h16a, out, 0, 0);
    // layer 3: write relu(h) to d_out
    lstm_layer_wmma<<<grid, 256, 0, stream>>>(
        x16, h16a, wih16 + 3 * WL, whh16 + 3 * WL,
        bih + 3 * GATES, bhh + 3 * GATES, cbuf, h16b, out, 0, 1);
}
